// EGNNFlowModel_82772609728933
// MI455X (gfx1250) — matrix operand order repaired and checked by reading, not compiled, not run on hardware
//
#include <hip/hip_runtime.h>
#include <cstdint>

typedef __attribute__((ext_vector_type(16))) _Float16 v16h;
typedef __attribute__((ext_vector_type(8)))  _Float16 v8h;
typedef __attribute__((ext_vector_type(8)))  float    v8f;

#define NLIG   16384
#define MPOC   32768
#define NTOT   (NLIG + MPOC)
#define ELIG   163840
#define EPOC   327680
#define ECROSS 131072
#define ETOT   (ELIG + EPOC + ECROSS)
#define KE1    224          // 2H+1+H = 193 padded to 7*32
#define NLAYERS 4
#define ETILES 4            // edge tile-groups per block (weight-staging amortization)
#define NTILES 4            // node tile-groups per block

__device__ __forceinline__ float silu_f(float v) { return v / (1.0f + __expf(-v)); }

// ---- async global -> LDS staging (CDNA5 GLOBAL_LOAD_ASYNC_TO_LDS_B128) ------
// Each lane moves 16B chunks straight into LDS; tracked by ASYNCcnt.
__device__ __forceinline__ void async_copy_g2l(const void* g, void* s, int n16,
                                               int tid, int nthreads) {
  uint32_t soff = (uint32_t)(uintptr_t)s;  // generic LDS ptr: low 32 bits = LDS offset
  const char* gp = (const char*)g;
  for (int i = tid; i < n16; i += nthreads) {
    uint32_t loff = soff + (uint32_t)i * 16u;
    const char* ga = gp + (size_t)i * 16;
    asm volatile("global_load_async_to_lds_b128 %0, %1, off"
                 :: "v"(loff), "v"(ga) : "memory");
  }
}
__device__ __forceinline__ void wait_async() {
  asm volatile("s_wait_asynccnt 0x0" ::: "memory");
}

// ---- WMMA fragment loaders (16-bit, 16x16x32) --------------------------------
// A (16xK row-major in LDS): lane L holds row M=L%16; half=L/16 selects K-halves:
//   a[0..7]  = K = kbase + 8*half + 0..7 ; a[8..15] = K = kbase + 16 + 8*half + 0..7
__device__ __forceinline__ v16h load_a16(const _Float16* row, int kbase, int half) {
  v16h a;
  v8h x0 = *(const v8h*)(row + kbase + 8 * half);
  v8h x1 = *(const v8h*)(row + kbase + 16 + 8 * half);
#pragma unroll
  for (int i = 0; i < 8; ++i) { a[i] = x0[i]; a[8 + i] = x1[i]; }
  return a;
}
// B from weights stored transposed [N][K]: lane L holds col N=L%16;
//   b[0..7]  = K = kbase + 8*half + 0..7 ; b[8..15] = K = kbase + 16 + 8*half + 0..7
__device__ __forceinline__ v16h load_b16(const _Float16* col, int kbase, int half) {
  v16h b;
  v8h x0 = *(const v8h*)(col + kbase + 8 * half);
  v8h x1 = *(const v8h*)(col + kbase + 16 + 8 * half);
#pragma unroll
  for (int i = 0; i < 8; ++i) { b[i] = x0[i]; b[8 + i] = x1[i]; }
  return b;
}

#define WMMA16(a, b, c) \
  __builtin_amdgcn_wmma_f32_16x16x32_f16(false, (a), false, (b), (short)0, (c), false, false)

// ---- weight prep: f32 [L][K][N] -> f16 transposed [L][N][Kpad] ---------------
__global__ void prep_w_kernel(const float* __restrict__ src, _Float16* __restrict__ dst,
                              int K, int Kpad, int N, int L) {
  int idx = blockIdx.x * blockDim.x + threadIdx.x;
  int tot = L * N * Kpad;
  if (idx >= tot) return;
  int k = idx % Kpad;
  int n = (idx / Kpad) % N;
  int l = idx / (Kpad * N);
  float v = (k < K) ? src[(size_t)l * K * N + (size_t)k * N + n] : 0.0f;
  dst[idx] = (_Float16)v;
}

// ---- node embeddings + coord init -------------------------------------------
__global__ void embed_kernel(const float* __restrict__ lig_x, const float* __restrict__ lig_h,
                             const float* __restrict__ poc_x, const float* __restrict__ poc_h,
                             const float* __restrict__ t,
                             const float* __restrict__ Wl, const float* __restrict__ bl,
                             const float* __restrict__ Wp, const float* __restrict__ bp,
                             float* __restrict__ h, float* __restrict__ x) {
  int idx = blockIdx.x * blockDim.x + threadIdx.x;
  if (idx >= NTOT * 64) return;
  int c = idx & 63, n = idx >> 6;
  float acc;
  if (n < NLIG) {
    acc = bl[c];
    const float* f = lig_h + (size_t)n * 17;
#pragma unroll
    for (int k = 0; k < 17; ++k) acc += f[k] * Wl[k * 64 + c];
    acc += t[0] * Wl[17 * 64 + c];
  } else {
    int j = n - NLIG;
    acc = bp[c];
    const float* f = poc_h + (size_t)j * 29;
#pragma unroll
    for (int k = 0; k < 29; ++k) acc += f[k] * Wp[k * 64 + c];
  }
  h[idx] = acc;
  if (c < 3) x[n * 3 + c] = (n < NLIG) ? lig_x[n * 3 + c] : poc_x[(size_t)(n - NLIG) * 3 + c];
}

// ---- edge-attr embeddings -> f16 --------------------------------------------
__global__ void eattr_kernel(const float* __restrict__ la, const float* __restrict__ pa,
                             const float* __restrict__ ca,
                             const float* __restrict__ Wl, const float* __restrict__ bl,
                             const float* __restrict__ Wp, const float* __restrict__ bp,
                             const float* __restrict__ Wc, const float* __restrict__ bc,
                             _Float16* __restrict__ eattr16) {
  int idx = blockIdx.x * blockDim.x + threadIdx.x;
  if (idx >= ETOT * 64) return;
  int c = idx & 63, e = idx >> 6;
  float acc;
  if (e < ELIG) {
    acc = bl[c];
    const float* f = la + (size_t)e * 6;
#pragma unroll
    for (int k = 0; k < 6; ++k) acc += f[k] * Wl[k * 64 + c];
  } else if (e < ELIG + EPOC) {
    int j = e - ELIG;
    acc = bp[c];
    const float* f = pa + (size_t)j * 16;
#pragma unroll
    for (int k = 0; k < 16; ++k) acc += f[k] * Wp[k * 64 + c];
  } else {
    int j = e - ELIG - EPOC;
    acc = bc[c];
    const float* f = ca + (size_t)j * 16;
#pragma unroll
    for (int k = 0; k < 16; ++k) acc += f[k] * Wc[k * 64 + c];
  }
  eattr16[idx] = (_Float16)acc;
}

// ---- edge kernel: phi_e + phi_x + scatter-add aggregates (WMMA) -------------
__global__ __launch_bounds__(128) void edge_kernel(
    const float* __restrict__ h, const float* __restrict__ x,
    const _Float16* __restrict__ eattr16,
    const int* __restrict__ lig_ei, const int* __restrict__ poc_ei,
    const int* __restrict__ cross_ei,
    const _Float16* __restrict__ We1t, const float* __restrict__ be1,
    const _Float16* __restrict__ We2t, const float* __restrict__ be2,
    const _Float16* __restrict__ Wx1t, const float* __restrict__ bx1,
    const _Float16* __restrict__ Wx2h, const float* __restrict__ bx2,
    float* __restrict__ msg_agg, float* __restrict__ coord_agg) {
  __shared__ __align__(16) _Float16 sW1[64 * KE1];   // phi_e W1^T [N][K]
  __shared__ __align__(16) _Float16 sW2[64 * 64];    // phi_e W2^T
  __shared__ __align__(16) _Float16 sX1[64 * 64];    // phi_x W1^T
  __shared__ __align__(16) _Float16 sX2[64];         // phi_x W2 (vector)
  __shared__ float sB1[64], sB2[64], sBx1[64];
  __shared__ __align__(16) _Float16 sA[4][16 * KE1]; // per-wave e_in panel
  __shared__ __align__(16) _Float16 sT[4][16 * 64];  // hidden (reused)
  __shared__ __align__(16) _Float16 sM[4][16 * 64];  // message m
  __shared__ float sDiff[4][16][3];
  __shared__ float sDist[4][16];
  __shared__ int sSrc[4][16], sDst[4][16];

  const int tid = threadIdx.x;
  // async weight staging straight into LDS (ASYNCcnt path)
  async_copy_g2l(We1t, sW1, 64 * KE1 * 2 / 16, tid, 128);
  async_copy_g2l(We2t, sW2, 64 * 64 * 2 / 16, tid, 128);
  async_copy_g2l(Wx1t, sX1, 64 * 64 * 2 / 16, tid, 128);
  async_copy_g2l(Wx2h, sX2, 64 * 2 / 16, tid, 128);
  if (tid < 64) { sB1[tid] = be1[tid]; sB2[tid] = be2[tid]; sBx1[tid] = bx1[tid]; }
  const float bx2v = bx2[0];
  wait_async();
  __syncthreads();

  const int w = tid >> 5, lane = tid & 31, half = lane >> 4, nl = lane & 15;

  for (int ti = 0; ti < ETILES; ++ti) {
    const int e0 = ((blockIdx.x * ETILES + ti) * 4 + w) * 16;

    if (half == 0) {  // per-row edge decode + geometry
      int e = e0 + nl, src, dst;
      if (e < ELIG) { src = lig_ei[e]; dst = lig_ei[ELIG + e]; }
      else if (e < ELIG + EPOC) { int j = e - ELIG; src = poc_ei[j] + NLIG; dst = poc_ei[EPOC + j] + NLIG; }
      else { int j = e - ELIG - EPOC; src = cross_ei[j] + NLIG; dst = cross_ei[ECROSS + j]; }
      float dx = x[src * 3 + 0] - x[dst * 3 + 0];
      float dy = x[src * 3 + 1] - x[dst * 3 + 1];
      float dz = x[src * 3 + 2] - x[dst * 3 + 2];
      sSrc[w][nl] = src; sDst[w][nl] = dst;
      sDiff[w][nl][0] = dx; sDiff[w][nl][1] = dy; sDiff[w][nl][2] = dz;
      sDist[w][nl] = (dx * dx + dy * dy + dz * dz) * 0.01f;
    }
    __builtin_amdgcn_wave_barrier();

    // stage e_in = [h[src] | h[dst] | dist | eattr | pad] as f16 rows
    for (int r = 0; r < 16; ++r) {
      const float* hs = h + (size_t)sSrc[w][r] * 64;
      const float* hd = h + (size_t)sDst[w][r] * 64;
      const _Float16* ea = eattr16 + (size_t)(e0 + r) * 64;
      float dv = sDist[w][r];
      for (int c = lane; c < KE1; c += 32) {
        _Float16 v;
        if (c < 64)        v = (_Float16)hs[c];
        else if (c < 128)  v = (_Float16)hd[c - 64];
        else if (c == 128) v = (_Float16)dv;
        else if (c < 193)  v = ea[c - 129];
        else               v = (_Float16)0.0f;
        sA[w][r * KE1 + c] = v;
      }
    }
    __builtin_amdgcn_wave_barrier();

    // prefetch next tile's edge-attr panel into L2 (global_prefetch_b8)
    if (ti + 1 < ETILES)
      __builtin_prefetch(eattr16 + (size_t)(e0 + 64) * 64 + (size_t)lane * 64, 0, 1);

    _Float16* myA = &sA[w][nl * KE1];
    _Float16* sTw = sT[w];
    _Float16* sMw = sM[w];

    // GEMM1: hidden = silu(e_in @ W1 + b1)     [16 x 224] @ [224 x 64]
#pragma unroll
    for (int nt = 0; nt < 4; ++nt) {
      int n = nt * 16 + nl;
      float bias = sB1[n];
      v8f c;
#pragma unroll
      for (int j = 0; j < 8; ++j) c[j] = bias;
      const _Float16* wcol = &sW1[n * KE1];
#pragma unroll
      for (int kt = 0; kt < KE1 / 32; ++kt) {
        v16h a = load_a16(myA, kt * 32, half);
        v16h b = load_b16(wcol, kt * 32, half);
        c = WMMA16(a, b, c);
      }
#pragma unroll
      for (int j = 0; j < 8; ++j) sTw[(j + 8 * half) * 64 + n] = (_Float16)silu_f(c[j]);
    }
    __builtin_amdgcn_wave_barrier();

    // GEMM2: m = silu(hidden @ W2 + b2), scatter msg_agg[dst]
#pragma unroll
    for (int nt = 0; nt < 4; ++nt) {
      int n = nt * 16 + nl;
      float bias = sB2[n];
      v8f c;
#pragma unroll
      for (int j = 0; j < 8; ++j) c[j] = bias;
      const _Float16* wcol = &sW2[n * 64];
#pragma unroll
      for (int kt = 0; kt < 2; ++kt) {
        v16h a = load_a16(&sTw[nl * 64], kt * 32, half);
        v16h b = load_b16(wcol, kt * 32, half);
        c = WMMA16(a, b, c);
      }
#pragma unroll
      for (int j = 0; j < 8; ++j) {
        int m = j + 8 * half;
        float mv = silu_f(c[j]);
        sMw[m * 64 + n] = (_Float16)mv;
        atomicAdd(&msg_agg[(size_t)sDst[w][m] * 64 + n], mv);
      }
    }
    __builtin_amdgcn_wave_barrier();

    // GEMM3: t1 = silu(m @ Wx1 + bx1)
#pragma unroll
    for (int nt = 0; nt < 4; ++nt) {
      int n = nt * 16 + nl;
      float bias = sBx1[n];
      v8f c;
#pragma unroll
      for (int j = 0; j < 8; ++j) c[j] = bias;
      const _Float16* wcol = &sX1[n * 64];
#pragma unroll
      for (int kt = 0; kt < 2; ++kt) {
        v16h a = load_a16(&sMw[nl * 64], kt * 32, half);
        v16h b = load_b16(wcol, kt * 32, half);
        c = WMMA16(a, b, c);
      }
#pragma unroll
      for (int j = 0; j < 8; ++j) sTw[(j + 8 * half) * 64 + n] = (_Float16)silu_f(c[j]);
    }
    __builtin_amdgcn_wave_barrier();

    // GEMV: cw = tanh(t1 . Wx2 + bx2); scatter coord_agg (ligand dst only)
    float s = 0.0f;
    {
      const _Float16* t1 = &sTw[nl * 64 + 32 * half];
      const _Float16* wv = &sX2[32 * half];
#pragma unroll
      for (int k = 0; k < 32; ++k) s += (float)t1[k] * (float)wv[k];
    }
    s += __shfl_xor(s, 16);
    if (half == 0) {
      int dst = sDst[w][nl];
      if (dst < NLIG) {
        float cw = tanhf(s + bx2v);
        atomicAdd(&coord_agg[dst * 3 + 0], sDiff[w][nl][0] * cw);
        atomicAdd(&coord_agg[dst * 3 + 1], sDiff[w][nl][1] * cw);
        atomicAdd(&coord_agg[dst * 3 + 2], sDiff[w][nl][2] * cw);
      }
    }
    __builtin_amdgcn_wave_barrier();
  }
}

// ---- node kernel: phi_h residual update + coord update (WMMA) ---------------
__global__ __launch_bounds__(128) void node_kernel(
    float* __restrict__ h, float* __restrict__ x,
    const float* __restrict__ msg_agg, const float* __restrict__ coord_agg,
    const _Float16* __restrict__ Wh1t, const float* __restrict__ bh1,
    const _Float16* __restrict__ Wh2t, const float* __restrict__ bh2) {
  __shared__ __align__(16) _Float16 sW1[64 * 128];
  __shared__ __align__(16) _Float16 sW2[64 * 64];
  __shared__ float sB1[64], sB2[64];
  __shared__ __align__(16) _Float16 sA[4][16 * 128];
  __shared__ __align__(16) _Float16 sT[4][16 * 64];

  const int tid = threadIdx.x;
  async_copy_g2l(Wh1t, sW1, 64 * 128 * 2 / 16, tid, 128);
  async_copy_g2l(Wh2t, sW2, 64 * 64 * 2 / 16, tid, 128);
  if (tid < 64) { sB1[tid] = bh1[tid]; sB2[tid] = bh2[tid]; }
  wait_async();
  __syncthreads();

  const int w = tid >> 5, lane = tid & 31, half = lane >> 4, nl = lane & 15;

  for (int ti = 0; ti < NTILES; ++ti) {
    const int n0 = ((blockIdx.x * NTILES + ti) * 4 + w) * 16;

    // stage [h | msg_agg] rows as f16
    for (int r = 0; r < 16; ++r) {
      const float* hp = h + (size_t)(n0 + r) * 64;
      const float* mp = msg_agg + (size_t)(n0 + r) * 64;
      for (int c = lane; c < 128; c += 32)
        sA[w][r * 128 + c] = (_Float16)((c < 64) ? hp[c] : mp[c - 64]);
    }
    __builtin_amdgcn_wave_barrier();

    _Float16* myA = &sA[w][nl * 128];
    _Float16* sTw = sT[w];

    // GEMM1: t = silu([h|msg] @ Wh1 + b1)     K=128
#pragma unroll
    for (int nt = 0; nt < 4; ++nt) {
      int n = nt * 16 + nl;
      float bias = sB1[n];
      v8f c;
#pragma unroll
      for (int j = 0; j < 8; ++j) c[j] = bias;
      const _Float16* wcol = &sW1[n * 128];
#pragma unroll
      for (int kt = 0; kt < 4; ++kt) {
        v16h a = load_a16(myA, kt * 32, half);
        v16h b = load_b16(wcol, kt * 32, half);
        c = WMMA16(a, b, c);
      }
#pragma unroll
      for (int j = 0; j < 8; ++j) sTw[(j + 8 * half) * 64 + n] = (_Float16)silu_f(c[j]);
    }
    __builtin_amdgcn_wave_barrier();

    // GEMM2: h += t @ Wh2 + b2
#pragma unroll
    for (int nt = 0; nt < 4; ++nt) {
      int n = nt * 16 + nl;
      float bias = sB2[n];
      v8f c;
#pragma unroll
      for (int j = 0; j < 8; ++j) c[j] = bias;
      const _Float16* wcol = &sW2[n * 64];
#pragma unroll
      for (int kt = 0; kt < 2; ++kt) {
        v16h a = load_a16(&sTw[nl * 64], kt * 32, half);
        v16h b = load_b16(wcol, kt * 32, half);
        c = WMMA16(a, b, c);
      }
#pragma unroll
      for (int j = 0; j < 8; ++j) {
        size_t o = (size_t)(n0 + j + 8 * half) * 64 + n;
        h[o] = h[o] + c[j];
      }
    }

    // x += coord_agg (ligand nodes only)
    if (half == 0) {
      int node = n0 + nl;
      if (node < NLIG) {
        x[node * 3 + 0] += coord_agg[node * 3 + 0];
        x[node * 3 + 1] += coord_agg[node * 3 + 1];
        x[node * 3 + 2] += coord_agg[node * 3 + 2];
      }
    }
    __builtin_amdgcn_wave_barrier();
  }
}

// ---- output head: scale * disp ----------------------------------------------
__global__ __launch_bounds__(64) void out_kernel(
    const float* __restrict__ h, const float* __restrict__ x,
    const float* __restrict__ lig_x,
    const float* __restrict__ W1, const float* __restrict__ b1,
    const float* __restrict__ W2, const float* __restrict__ b2,
    float* __restrict__ out) {
  __shared__ float red[64];
  int n = blockIdx.x, c = threadIdx.x;
  const float* hn = h + (size_t)n * 64;
  float acc = b1[c];
#pragma unroll
  for (int k = 0; k < 64; ++k) acc += hn[k] * W1[k * 64 + c];
  red[c] = silu_f(acc) * W2[c];
  __syncthreads();
  for (int sft = 32; sft > 0; sft >>= 1) {
    if (c < sft) red[c] += red[c + sft];
    __syncthreads();
  }
  if (c < 3) {
    float scale = red[0] + b2[0];
    out[n * 3 + c] = scale * (x[n * 3 + c] - lig_x[n * 3 + c]);
  }
}

// ---- host launch -------------------------------------------------------------
extern "C" void kernel_launch(void* const* d_in, const int* in_sizes, int n_in,
                              void* d_out, int out_size, void* d_ws, size_t ws_size,
                              hipStream_t stream) {
  (void)in_sizes; (void)n_in; (void)out_size; (void)ws_size;
  const float* lig_x   = (const float*)d_in[0];
  const float* lig_h   = (const float*)d_in[1];
  const float* poc_x   = (const float*)d_in[2];
  const float* poc_h   = (const float*)d_in[3];
  const int*   lig_ei  = (const int*)d_in[4];
  const float* lig_ea  = (const float*)d_in[5];
  const int*   poc_ei  = (const int*)d_in[6];
  const float* poc_ea  = (const float*)d_in[7];
  const int*   cross_ei= (const int*)d_in[8];
  const float* cross_ea= (const float*)d_in[9];
  const float* t       = (const float*)d_in[10];
  const float* lig_emb_W = (const float*)d_in[11]; const float* lig_emb_b = (const float*)d_in[12];
  const float* poc_emb_W = (const float*)d_in[13]; const float* poc_emb_b = (const float*)d_in[14];
  const float* lig_ee_W  = (const float*)d_in[15]; const float* lig_ee_b  = (const float*)d_in[16];
  const float* poc_ee_W  = (const float*)d_in[17]; const float* poc_ee_b  = (const float*)d_in[18];
  const float* cross_ee_W= (const float*)d_in[19]; const float* cross_ee_b= (const float*)d_in[20];
  const float* phi_e_W1 = (const float*)d_in[21]; const float* phi_e_b1 = (const float*)d_in[22];
  const float* phi_e_W2 = (const float*)d_in[23]; const float* phi_e_b2 = (const float*)d_in[24];
  const float* phi_x_W1 = (const float*)d_in[25]; const float* phi_x_b1 = (const float*)d_in[26];
  const float* phi_x_W2 = (const float*)d_in[27]; const float* phi_x_b2 = (const float*)d_in[28];
  const float* phi_h_W1 = (const float*)d_in[29]; const float* phi_h_b1 = (const float*)d_in[30];
  const float* phi_h_W2 = (const float*)d_in[31]; const float* phi_h_b2 = (const float*)d_in[32];
  const float* out_W1 = (const float*)d_in[33]; const float* out_b1 = (const float*)d_in[34];
  const float* out_W2 = (const float*)d_in[35]; const float* out_b2 = (const float*)d_in[36];

  char* p = (char*)d_ws;
  auto alloc = [&](size_t bytes) { char* r = p; p += (bytes + 255) & ~(size_t)255; return r; };
  float*     h       = (float*)alloc((size_t)NTOT * 64 * 4);
  float*     x       = (float*)alloc((size_t)NTOT * 3 * 4);
  float*     msg     = (float*)alloc((size_t)NTOT * 64 * 4);
  float*     coord   = (float*)alloc((size_t)NTOT * 3 * 4);
  _Float16*  eattr16 = (_Float16*)alloc((size_t)ETOT * 64 * 2);
  _Float16*  We1t    = (_Float16*)alloc((size_t)NLAYERS * 64 * KE1 * 2);
  _Float16*  We2t    = (_Float16*)alloc((size_t)NLAYERS * 64 * 64 * 2);
  _Float16*  Wx1t    = (_Float16*)alloc((size_t)NLAYERS * 64 * 64 * 2);
  _Float16*  Wx2h    = (_Float16*)alloc((size_t)NLAYERS * 64 * 2);
  _Float16*  Wh1t    = (_Float16*)alloc((size_t)NLAYERS * 64 * 128 * 2);
  _Float16*  Wh2t    = (_Float16*)alloc((size_t)NLAYERS * 64 * 64 * 2);

  auto prep = [&](const float* src, _Float16* dst, int K, int Kpad, int N) {
    int tot = NLAYERS * N * Kpad;
    prep_w_kernel<<<(tot + 255) / 256, 256, 0, stream>>>(src, dst, K, Kpad, N, NLAYERS);
  };
  prep(phi_e_W1, We1t, 193, KE1, 64);
  prep(phi_e_W2, We2t, 64, 64, 64);
  prep(phi_x_W1, Wx1t, 64, 64, 64);
  prep(phi_x_W2, Wx2h, 64, 64, 1);   // column vector [L][1][64]
  prep(phi_h_W1, Wh1t, 128, 128, 64);
  prep(phi_h_W2, Wh2t, 64, 64, 64);

  embed_kernel<<<(NTOT * 64) / 256, 256, 0, stream>>>(
      lig_x, lig_h, poc_x, poc_h, t, lig_emb_W, lig_emb_b, poc_emb_W, poc_emb_b, h, x);
  eattr_kernel<<<(ETOT * 64) / 256, 256, 0, stream>>>(
      lig_ea, poc_ea, cross_ea, lig_ee_W, lig_ee_b, poc_ee_W, poc_ee_b,
      cross_ee_W, cross_ee_b, eattr16);

  for (int l = 0; l < NLAYERS; ++l) {
    hipMemsetAsync(msg, 0, (size_t)NTOT * 64 * 4, stream);
    hipMemsetAsync(coord, 0, (size_t)NTOT * 3 * 4, stream);
    edge_kernel<<<ETOT / (64 * ETILES), 128, 0, stream>>>(
        h, x, eattr16, lig_ei, poc_ei, cross_ei,
        We1t + (size_t)l * 64 * KE1, phi_e_b1 + l * 64,
        We2t + (size_t)l * 64 * 64, phi_e_b2 + l * 64,
        Wx1t + (size_t)l * 64 * 64, phi_x_b1 + l * 64,
        Wx2h + (size_t)l * 64, phi_x_b2 + l,
        msg, coord);
    node_kernel<<<NTOT / (64 * NTILES), 128, 0, stream>>>(
        h, x, msg, coord,
        Wh1t + (size_t)l * 64 * 128, phi_h_b1 + l * 64,
        Wh2t + (size_t)l * 64 * 64, phi_h_b2 + l * 64);
  }

  out_kernel<<<NLIG, 64, 0, stream>>>(h, x, lig_x, out_W1, out_b1, out_W2, out_b2,
                                      (float*)d_out);
}
